// BasicDecBlk_39135742001993
// MI455X (gfx1250) — matrix operand order repaired and checked by reading, not compiled, not run on hardware
//
#include <hip/hip_runtime.h>
#include <hip/hip_bf16.h>
#include <cstdint>
#include <cstddef>

typedef __attribute__((ext_vector_type(4)))  _Float16 v4h;
typedef __attribute__((ext_vector_type(8)))  _Float16 v8h;
typedef __attribute__((ext_vector_type(16))) _Float16 v16h;
typedef __attribute__((ext_vector_type(8)))  float    v8f;

#define BB   2
#define HH   96
#define WW   96
#define NPIX (BB*HH*WW)      /* 18432 pixels */
#define EPSbn 1e-5f

// Async global->LDS 16B copy (CDNA5 GLOBAL_LOAD_ASYNC_TO_LDS_B128, ASYNCcnt).
// lds_off: byte offset in wave LDS space (low 32 bits of flat shared addr).
__device__ __forceinline__ void async_copy_b128(uint32_t lds_off, const void* gptr) {
  asm volatile("global_load_async_to_lds_b128 %0, %1, off"
               :: "v"(lds_off), "v"((unsigned long long)(uintptr_t)gptr)
               : "memory");
}
__device__ __forceinline__ void async_wait0() {
  asm volatile("s_wait_asynccnt 0x0" ::: "memory");
}

// ---------------------------------------------------------------------------
// Pack f32 weights [O][Kd] (OIHW-flat) into f16 O-major [Opad][Kd], zero rows.
// ---------------------------------------------------------------------------
__global__ __launch_bounds__(256) void pack_w_kernel(
    const float* __restrict__ w, _Float16* __restrict__ wt,
    int O, int Opad, int Kd) {
  int idx = blockIdx.x * blockDim.x + threadIdx.x;
  int tot = Kd * Opad;
  if (idx >= tot) return;
  int o = idx / Kd, kd = idx % Kd;
  float v = (o < O) ? w[(size_t)o * Kd + kd] : 0.f;
  wt[idx] = (_Float16)v;
}

// ---------------------------------------------------------------------------
// im2col: src is NCHW f32 (nchw=1) or pixel-major [N][64] f32 (nchw=0).
// col[n][(c*k+dy)*k+dx] f16, zero outside bounds. C fixed at 64.
// ---------------------------------------------------------------------------
__global__ __launch_bounds__(256) void im2col_kernel(
    const float* __restrict__ src, _Float16* __restrict__ col,
    int k, int pad, int nchw) {
  int kk2 = k * k;
  int CKK = 64 * kk2;
  long idx = (long)blockIdx.x * blockDim.x + threadIdx.x;
  long tot = (long)NPIX * CKK;
  if (idx >= tot) return;
  int n  = (int)(idx / CKK);
  int kd = (int)(idx % CKK);
  int c = kd / kk2, rem = kd % kk2, dy = rem / k, dx = rem % k;
  int b = n / (HH * WW), r = n % (HH * WW), y = r / WW, x = r % WW;
  int yy = y + dy - pad, xx = x + dx - pad;
  float v = 0.f;
  if (yy >= 0 && yy < HH && xx >= 0 && xx < WW) {
    v = nchw ? src[(((size_t)b * 64 + c) * HH + yy) * WW + xx]
             : src[((size_t)(b * HH + yy) * WW + xx) * 64 + c];
  }
  col[idx] = (_Float16)v;
}

// ---------------------------------------------------------------------------
// WMMA GEMM: C[N x O] = X[N x Kd](f16) * Wt[Opad x Kd](f16, O-major), f32 acc.
// Block: 256 thr = 8 waves, tile 128 pixels x 64 outch; wave tile 32x32
// (2x2 sub-tiles -> 4 WMMAs per K-step off 2 A + 2 B fragments).
// Tiles staged via GLOBAL_LOAD_ASYNC_TO_LDS_B128 (ASYNCcnt).
// Epilogue: +bias, BN(scale/shift), act(0 none,1 relu,2 2*sigmoid),
// out_mode: 0 f32 [n*ldd+coloff+o], 1 f16 same, 2 f32 NCHW scatter (O ch).
// ---------------------------------------------------------------------------
__global__ __launch_bounds__(256) void gemm_wmma_kernel(
    const _Float16* __restrict__ X, const _Float16* __restrict__ Wt,
    int Kd, int Opad, int O,
    const float* __restrict__ bias,
    const float* __restrict__ bng, const float* __restrict__ bnb,
    const float* __restrict__ bnm, const float* __restrict__ bnv,
    int act, int out_mode,
    float* __restrict__ of32, _Float16* __restrict__ of16,
    int ldd, int coloff) {
  __shared__ __align__(16) _Float16 Xs[128 * 40];
  __shared__ __align__(16) _Float16 Ws[64 * 40];
  const int t    = threadIdx.x;
  const int lane = t & 31;
  const int wid  = t >> 5;
  const int wm   = wid & 3;       // 4 pixel tiles of 32
  const int wn   = wid >> 2;      // 2 outch tiles of 32
  const int n0   = blockIdx.x * 128;
  const int o0   = blockIdx.y * 64;
  const int g16  = lane >> 4;     // lane-half selector
  const int mrow = lane & 15;

  const uint32_t ldsX = (uint32_t)(uintptr_t)(&Xs[0]);
  const uint32_t ldsW = (uint32_t)(uintptr_t)(&Ws[0]);
  // X tile loader coords: 128 rows x 4 chunks of 8 halves; 2 chunks/thread.
  const int xr = t >> 1, xseg = (t & 1) * 2;
  // W tile loader coords: 64 rows x 4 chunks; 1 chunk/thread.
  const int wo = t >> 2, wseg = t & 3;

  v8f acc00 = {}, acc01 = {}, acc10 = {}, acc11 = {};
  for (int kb = 0; kb < Kd; kb += 32) {
    // async-stage X tile (128x32 halves) and W tile (64x32 halves)
    async_copy_b128(ldsX + (uint32_t)(xr * 40 + xseg * 8) * 2,
                    X + (size_t)(n0 + xr) * Kd + kb + xseg * 8);
    async_copy_b128(ldsX + (uint32_t)(xr * 40 + (xseg + 1) * 8) * 2,
                    X + (size_t)(n0 + xr) * Kd + kb + (xseg + 1) * 8);
    async_copy_b128(ldsW + (uint32_t)(wo * 40 + wseg * 8) * 2,
                    Wt + (size_t)(o0 + wo) * Kd + kb + wseg * 8);
    async_wait0();
    __syncthreads();

    // Fragments (ISA 16-bit layouts):
    // A: M=lane&15, K(i) = (i>>3)*16 + g16*8 + (i&7)
    // B: N=lane&15, K(i) = (i>>3)*16 + g16*8 + (i&7)
    union { v16h v; v8h h[2]; } a0, a1, b0, b1;
    {
      int r0 = (wm * 32 + mrow) * 40;
      int r1 = (wm * 32 + 16 + mrow) * 40;
      a0.h[0] = *(const v8h*)(&Xs[r0 + g16 * 8]);
      a0.h[1] = *(const v8h*)(&Xs[r0 + 16 + g16 * 8]);
      a1.h[0] = *(const v8h*)(&Xs[r1 + g16 * 8]);
      a1.h[1] = *(const v8h*)(&Xs[r1 + 16 + g16 * 8]);
      int c0 = (wn * 32 + mrow) * 40;
      int c1 = (wn * 32 + 16 + mrow) * 40;
      b0.h[0] = *(const v8h*)(&Ws[c0 + g16 * 8]);
      b0.h[1] = *(const v8h*)(&Ws[c0 + 16 + g16 * 8]);
      b1.h[0] = *(const v8h*)(&Ws[c1 + g16 * 8]);
      b1.h[1] = *(const v8h*)(&Ws[c1 + 16 + g16 * 8]);
    }
    acc00 = __builtin_amdgcn_wmma_f32_16x16x32_f16(false, a0.v, false, b0.v, (short)0, acc00, false, false);
    acc01 = __builtin_amdgcn_wmma_f32_16x16x32_f16(false, a0.v, false, b1.v, (short)0, acc01, false, false);
    acc10 = __builtin_amdgcn_wmma_f32_16x16x32_f16(false, a1.v, false, b0.v, (short)0, acc10, false, false);
    acc11 = __builtin_amdgcn_wmma_f32_16x16x32_f16(false, a1.v, false, b1.v, (short)0, acc11, false, false);
    __syncthreads();
  }

  // Epilogue: acc[r] -> (M = r + g16*8, N = lane&15)
  for (int ni = 0; ni < 2; ++ni) {
    int oN = o0 + wn * 32 + ni * 16 + (lane & 15);
    if (oN >= O) continue;
    float sc = 1.f, sh = 0.f;
    if (bng) {
      float s = bng[oN] * rsqrtf(bnv[oN] + EPSbn);
      sc = s; sh = bnb[oN] - bnm[oN] * s;
    }
    float bs = bias ? bias[oN] : 0.f;
    for (int mi = 0; mi < 2; ++mi) {
      const v8f* ap = (mi == 0) ? (ni == 0 ? &acc00 : &acc01)
                                : (ni == 0 ? &acc10 : &acc11);
      for (int r = 0; r < 8; ++r) {
        int n = n0 + wm * 32 + mi * 16 + r + (g16 ? 8 : 0);
        float v = ((*ap)[r] + bs) * sc + sh;
        if (act == 1)      v = v > 0.f ? v : 0.f;
        else if (act == 2) v = 2.f / (1.f + __expf(-v));
        if (out_mode == 0)      of32[(size_t)n * ldd + coloff + oN] = v;
        else if (out_mode == 1) of16[(size_t)n * ldd + coloff + oN] = (_Float16)v;
        else {
          int bI = n / (HH * WW), rr = n % (HH * WW), y = rr / WW, x = rr % WW;
          of32[(((size_t)bI * O + oN) * HH + y) * WW + x] = v;
        }
      }
    }
  }
}

// ---------------------------------------------------------------------------
// Deformable sampling: builds S[n][c*K + tap] = bilinear(h, base+offset)*mask
// (offset channels alternate dy,dx; clip indices, zero weight outside).
// ---------------------------------------------------------------------------
__global__ __launch_bounds__(256) void build_s_kernel(
    const float* __restrict__ h,
    const float* __restrict__ off, int ldo,
    const float* __restrict__ mod, int ldm,
    _Float16* __restrict__ S, int k, int pad) {
  int K = k * k;
  long idx = (long)blockIdx.x * blockDim.x + threadIdx.x;
  long tot = (long)NPIX * K;
  if (idx >= tot) return;
  int n = (int)(idx / K), tap = (int)(idx % K);
  int b = n / (HH * WW), r = n % (HH * WW), y = r / WW, x = r % WW;
  int dy = tap / k, dx = tap % k;
  float ys = (float)(y + dy - pad) + off[(size_t)n * ldo + 2 * tap];
  float xs = (float)(x + dx - pad) + off[(size_t)n * ldo + 2 * tap + 1];
  float m  = mod[(size_t)n * ldm + tap];
  float y0 = floorf(ys), x0 = floorf(xs);
  const float* p[4]; float wt[4];
  int q = 0;
  for (int jy = 0; jy < 2; ++jy)
    for (int jx = 0; jx < 2; ++jx) {
      float yi = y0 + jy, xi = x0 + jx;
      float wgt = (1.f - fabsf(ys - yi)) * (1.f - fabsf(xs - xi));
      bool valid = (yi >= 0.f) && (yi < (float)HH) &&
                   (xi >= 0.f) && (xi < (float)WW);
      int yc = (int)fminf(fmaxf(yi, 0.f), (float)(HH - 1));
      int xc = (int)fminf(fmaxf(xi, 0.f), (float)(WW - 1));
      p[q]  = h + ((size_t)(b * HH + yc) * WW + xc) * 64;
      wt[q] = valid ? wgt : 0.f;
      ++q;
    }
  size_t sbase = (size_t)n * (64 * K) + tap;
  for (int c = 0; c < 64; ++c) {
    float v = wt[0]*p[0][c] + wt[1]*p[1][c] + wt[2]*p[2][c] + wt[3]*p[3][c];
    S[sbase + (size_t)c * K] = (_Float16)(v * m);
  }
}

// ---------------------------------------------------------------------------
// Global average pool over pixels: g[b*64+c] = mean(h[b,:,:,c])
// ---------------------------------------------------------------------------
__global__ __launch_bounds__(256) void gap_reduce_kernel(
    const float* __restrict__ h, float* __restrict__ g) {
  __shared__ float red[256];
  int bc = blockIdx.x;             // 0..127
  int b = bc >> 6, c = bc & 63;
  float s = 0.f;
  for (int p = threadIdx.x; p < HH * WW; p += 256)
    s += h[((size_t)b * HH * WW + p) * 64 + c];
  red[threadIdx.x] = s;
  __syncthreads();
  for (int st = 128; st > 0; st >>= 1) {
    if (threadIdx.x < st) red[threadIdx.x] += red[threadIdx.x + st];
    __syncthreads();
  }
  if (threadIdx.x == 0) g[bc] = red[0] / (float)(HH * WW);
}

__global__ __launch_bounds__(256) void gap_fc_kernel(
    const float* __restrict__ g, const float* __restrict__ w,
    const float* __restrict__ bg, const float* __restrict__ bb,
    const float* __restrict__ bm, const float* __restrict__ bv,
    float* __restrict__ gout) {
  int i = blockIdx.x * blockDim.x + threadIdx.x;
  if (i >= 2 * 256) return;
  int b = i >> 8, o = i & 255;
  float s = 0.f;
  for (int c = 0; c < 64; ++c) s += g[b * 64 + c] * w[o * 64 + c];
  float sc = bg[o] * rsqrtf(bv[o] + EPSbn);
  float v = s * sc + (bb[o] - bm[o] * sc);
  gout[i] = v > 0.f ? v : 0.f;
}

__global__ __launch_bounds__(256) void gap_bcast_kernel(
    const float* __restrict__ gout, _Float16* __restrict__ cat) {
  long idx = (long)blockIdx.x * blockDim.x + threadIdx.x;
  if (idx >= (long)NPIX * 256) return;
  int n = (int)(idx >> 8), o = (int)(idx & 255);
  int b = n / (HH * WW);
  cat[(size_t)n * 1280 + 1024 + o] = (_Float16)gout[b * 256 + o];
}

// ---------------------------------------------------------------------------
extern "C" void kernel_launch(void* const* d_in, const int* in_sizes, int n_in,
                              void* d_out, int out_size, void* d_ws, size_t ws_size,
                              hipStream_t stream) {
  (void)in_sizes; (void)n_in; (void)out_size; (void)ws_size;

  // ---- input unpack (setup_inputs() dict order, depth-first) ----
  const float* x_in      = (const float*)d_in[0];
  const float* conv_in_w = (const float*)d_in[1];
  const float* conv_in_b = (const float*)d_in[2];
  const float* bn_in_g = (const float*)d_in[3], *bn_in_b = (const float*)d_in[4];
  const float* bn_in_m = (const float*)d_in[5], *bn_in_v = (const float*)d_in[6];
  // branches: aspp1@7, aspp_d1@16, aspp_d3@25, aspp_d7@34 (each 9 entries)
  const float* gap_w   = (const float*)d_in[43];
  const float* gbn_g = (const float*)d_in[44], *gbn_b = (const float*)d_in[45];
  const float* gbn_m = (const float*)d_in[46], *gbn_v = (const float*)d_in[47];
  const float* conv1_w = (const float*)d_in[48];
  const float* bn1_g = (const float*)d_in[49], *bn1_b = (const float*)d_in[50];
  const float* bn1_m = (const float*)d_in[51], *bn1_v = (const float*)d_in[52];
  const float* conv_out_w = (const float*)d_in[53];
  const float* conv_out_b = (const float*)d_in[54];
  const float* bno_g = (const float*)d_in[55], *bno_b = (const float*)d_in[56];
  const float* bno_m = (const float*)d_in[57], *bno_v = (const float*)d_in[58];

  // ---- workspace partition (~188 MB) ----
  char* ws = (char*)d_ws;
  size_t off = 0;
  auto carve = [&](size_t bytes) -> char* {
    char* p = ws + off;
    off += (bytes + 255) & ~(size_t)255;
    return p;
  };
  float*    h_buf  = (float*)   carve((size_t)NPIX * 64 * 4);
  float*    a_buf  = (float*)   carve((size_t)NPIX * 64 * 4);
  float*    g_buf  = (float*)   carve(128 * 4);
  float*    go_buf = (float*)   carve(512 * 4);
  _Float16* wt_buf = (_Float16*)carve((size_t)3136 * 256 * 2);
  float*    off_b  = (float*)   carve((size_t)NPIX * 128 * 4);
  float*    mod_b  = (float*)   carve((size_t)NPIX * 64 * 4);
  _Float16* cat_b  = (_Float16*)carve((size_t)NPIX * 1280 * 2);
  _Float16* col_b  = (_Float16*)carve((size_t)NPIX * 3136 * 2);

  auto gemm = [&](const _Float16* X, int Kd, const float* w, int O,
                  const float* bias, const float* bg, const float* bb,
                  const float* bm, const float* bv, int act, int out_mode,
                  float* of32, _Float16* of16, int ldd, int coloff) {
    int Opad = (O + 63) & ~63;
    int tot = Kd * Opad;
    pack_w_kernel<<<(tot + 255) / 256, 256, 0, stream>>>(w, wt_buf, O, Opad, Kd);
    dim3 grid(NPIX / 128, Opad / 64);
    gemm_wmma_kernel<<<grid, 256, 0, stream>>>(X, wt_buf, Kd, Opad, O, bias,
        bg, bb, bm, bv, act, out_mode, of32, of16, ldd, coloff);
  };

  // ---- conv_in (3x3, 64->64) + BN + ReLU -> h (pixel-major [N][64] f32) ----
  {
    long tot = (long)NPIX * 576;
    im2col_kernel<<<(tot + 255) / 256, 256, 0, stream>>>(x_in, col_b, 3, 1, 1);
    gemm(col_b, 576, conv_in_w, 64, conv_in_b, bn_in_g, bn_in_b, bn_in_m, bn_in_v,
         /*act*/1, /*mode*/0, h_buf, nullptr, 64, 0);
  }

  // ---- GAP branch -> cat columns [1024,1280) ----
  gap_reduce_kernel<<<128, 256, 0, stream>>>(h_buf, g_buf);
  gap_fc_kernel<<<2, 256, 0, stream>>>(g_buf, gap_w, gbn_g, gbn_b, gbn_m, gbn_v, go_buf);
  {
    long tot = (long)NPIX * 256;
    gap_bcast_kernel<<<(tot + 255) / 256, 256, 0, stream>>>(go_buf, cat_b);
  }

  // ---- deformable branches ----
  struct Br { int k, pad, base, coloff; };
  const Br br[4] = { {1, 0, 7, 0}, {1, 0, 16, 256}, {3, 1, 25, 512}, {7, 3, 34, 768} };
  for (int bi = 0; bi < 4; ++bi) {
    int k = br[bi].k, pad = br[bi].pad, base = br[bi].base;
    int K = k * k, CKK = 64 * K;
    const float* p_off_w = (const float*)d_in[base + 0];
    const float* p_off_b = (const float*)d_in[base + 1];
    const float* p_mod_w = (const float*)d_in[base + 2];
    const float* p_mod_b = (const float*)d_in[base + 3];
    const float* p_reg_w = (const float*)d_in[base + 4];
    const float* p_bg = (const float*)d_in[base + 5];
    const float* p_bb = (const float*)d_in[base + 6];
    const float* p_bm = (const float*)d_in[base + 7];
    const float* p_bv = (const float*)d_in[base + 8];

    long tot = (long)NPIX * CKK;
    im2col_kernel<<<(tot + 255) / 256, 256, 0, stream>>>(h_buf, col_b, k, pad, 0);

    int Ooff = 2 * K, OpadOff = (Ooff + 63) & ~63;
    gemm(col_b, CKK, p_off_w, Ooff, p_off_b, nullptr, nullptr, nullptr, nullptr,
         /*act*/0, /*mode*/0, off_b, nullptr, OpadOff, 0);

    int Omod = K, OpadMod = (Omod + 63) & ~63;
    gemm(col_b, CKK, p_mod_w, Omod, p_mod_b, nullptr, nullptr, nullptr, nullptr,
         /*act*/2, /*mode*/0, mod_b, nullptr, OpadMod, 0);

    long ts = (long)NPIX * K;
    build_s_kernel<<<(ts + 255) / 256, 256, 0, stream>>>(
        h_buf, off_b, OpadOff, mod_b, OpadMod, col_b, k, pad);

    gemm(col_b, CKK, p_reg_w, 256, nullptr, p_bg, p_bb, p_bm, p_bv,
         /*act*/1, /*mode*/1, nullptr, cat_b, 1280, br[bi].coloff);
  }

  // ---- conv1 (1x1, 1280->64) + BN + ReLU -> a ----
  gemm(cat_b, 1280, conv1_w, 64, nullptr, bn1_g, bn1_b, bn1_m, bn1_v,
       /*act*/1, /*mode*/0, a_buf, nullptr, 64, 0);

  // ---- conv_out (3x3, 64->64) + bias + BN -> d_out (NCHW f32) ----
  {
    long tot = (long)NPIX * 576;
    im2col_kernel<<<(tot + 255) / 256, 256, 0, stream>>>(a_buf, col_b, 3, 1, 0);
    gemm(col_b, 576, conv_out_w, 64, conv_out_b, bno_g, bno_b, bno_m, bno_v,
         /*act*/0, /*mode*/2, (float*)d_out, nullptr, 0, 0);
  }
}
